// BiLSTMCRF_13984413516542
// MI455X (gfx1250) — compile-verified, matrix operand order checked by reading, not run on hardware
//
#include <hip/hip_runtime.h>
#include <hip/hip_bf16.h>

// BiLSTM-CRF for MI455X (gfx1250, wave32).
// Matmuls: v_wmma_f32_16x16x32_bf16. Big GEMM stages tiles via the Tensor
// Data Mover (tensor_load_to_lds + s_wait_tensorcnt) when available.

#ifndef __has_builtin
#define __has_builtin(x) 0
#endif
#if __has_builtin(__builtin_amdgcn_tensor_load_to_lds)
#define HAVE_TDM 1
#else
#define HAVE_TDM 0
#endif

typedef __attribute__((ext_vector_type(16))) __bf16 v16bf;
typedef __attribute__((ext_vector_type(8)))  __bf16 v8bf;
typedef __attribute__((ext_vector_type(8)))  float  v8f;
typedef __attribute__((ext_vector_type(4)))  unsigned int u32x4;
typedef __attribute__((ext_vector_type(4)))  int i32x4;
typedef __attribute__((ext_vector_type(8)))  int i32x8;

__device__ __forceinline__ v8f wmma_bf16(v16bf a, v16bf b, v8f c) {
  return __builtin_amdgcn_wmma_f32_16x16x32_bf16(false, a, false, b, (short)0, c,
                                                 false, false);
}

// Fragment loader per ISA 7.12.2 (16-bit A/B 16x32): lane l holds row l&15;
// lanes<16: K 0..7 & 16..23, lanes>=16: K 8..15 & 24..31. -> 2x b128 per lane.
__device__ __forceinline__ v16bf frag_ld(const __bf16* p, int ld) {
  const int lane = threadIdx.x & 31;
  const int row  = lane & 15;
  const int kh   = (lane >> 4) << 3;
  const __bf16* r = p + (size_t)row * ld + kh;
  v16bf a;
#pragma unroll
  for (int i = 0; i < 8; ++i) a[i] = r[i];
#pragma unroll
  for (int i = 0; i < 8; ++i) a[8 + i] = r[16 + i];
  return a;
}

#if HAVE_TDM
// TDM 2D tile load: tile_rows x 32 bf16, global row stride = ld elements,
// LDS rows padded +16B (pad_interval=16 dwords, pad_amount=4 dwords) so the
// staged row stride is 40 elements -> conflict-free ds_load_b128 frag reads.
__device__ __forceinline__ void tdm_load_2d(const __bf16* lds_dst,
                                            const __bf16* gsrc,
                                            unsigned tile_rows, unsigned ld) {
  const unsigned long long ga = (unsigned long long)(const void*)gsrc;
  u32x4 g0;
  g0[0] = 1u;                                             // count=1 (valid D#)
  g0[1] = (unsigned)(unsigned long long)(const void*)lds_dst; // LDS byte addr
  g0[2] = (unsigned)(ga & 0xffffffffull);
  g0[3] = (unsigned)((ga >> 32) & 0x1ffffffull) | 0x80000000u; // type=2
  i32x8 g1;
  g1[0] = (int)((1u << 16)            // data_size = 2 bytes
              | (1u << 20)            // pad_enable
              | (3u << 22)            // pad_interval = 16 dwords (one row)
              | (3u << 25));          // pad_amount   = 4 dwords (16B)
  g1[1] = (int)((32u & 0xffffu) << 16);                   // tensor_dim0 = 32
  g1[2] = (int)((tile_rows & 0xffffu) << 16);             // tensor_dim1
  g1[3] = (int)((32u & 0xffffu) << 16);                   // tile_dim0 = 32
  g1[4] = (int)(tile_rows & 0xffffu);                     // tile_dim1
  g1[5] = (int)ld;                                        // dim0 stride lo
  g1[6] = 0;
  g1[7] = 0;
  i32x4 z4 = {0, 0, 0, 0};
#if __has_include(<hip/amd_detail/amd_gfx1250_TDM.h>)
  i32x8 z8 = {0, 0, 0, 0, 0, 0, 0, 0};
  __builtin_amdgcn_tensor_load_to_lds(g0, g1, z4, z4, z8, 0);
#else
  __builtin_amdgcn_tensor_load_to_lds(g0, g1, z4, z4, 0);
#endif
}
__device__ __forceinline__ void wait_tensor0() {
#if __has_builtin(__builtin_amdgcn_s_wait_tensorcnt)
  __builtin_amdgcn_s_wait_tensorcnt((short)0);
#endif
}
#endif // HAVE_TDM

// ---------------- conversion kernels ----------------
__global__ void k_cvt_bf16(const float* __restrict__ src, __bf16* __restrict__ dst, int n) {
  int i = blockIdx.x * blockDim.x + threadIdx.x;
  int stride = gridDim.x * blockDim.x;
  for (; i < n; i += stride) dst[i] = (__bf16)src[i];
}

__global__ void k_pack_fcw(const float* __restrict__ fcw, __bf16* __restrict__ dst) {
  int i = blockIdx.x * blockDim.x + threadIdx.x;
  if (i >= 16 * 512) return;
  int n = i >> 9, k = i & 511;
  dst[i] = (__bf16)((n < 9) ? fcw[n * 512 + k] : 0.f);
}

// ---------------- input GEMM: gx = feats @ w_ih^T + b_ih + b_hh -------------
// M = 32768, K = 768, N = 1024; block tile 128x128, wave tile 16x128.
// Output layout gx[dir][l][g][64 b] (bf16) so the recurrent kernel reads one
// b128 per accumulator tile.
#define SROW 40   // staged LDS row stride (32 data + 8 pad bf16)
__global__ void __launch_bounds__(256)
k_gemm_gx(const __bf16* __restrict__ featsb,
          const __bf16* __restrict__ wih_f, const __bf16* __restrict__ wih_b,
          const float* __restrict__ bih_f, const float* __restrict__ bhh_f,
          const float* __restrict__ bih_b, const float* __restrict__ bhh_b,
          __bf16* __restrict__ gx) {
  const int dir = blockIdx.z;
  const __bf16* wih = dir ? wih_b : wih_f;
  const float* bi = dir ? bih_b : bih_f;
  const float* bh = dir ? bhh_b : bhh_f;
  __bf16* gout = gx + (size_t)dir * 512 * 64 * 1024;

  const int wave = threadIdx.x >> 5;
  const int lane = threadIdx.x & 31;
  const int m0blk = blockIdx.x * 128;
  const int m0 = m0blk + wave * 16;
  const int g0 = blockIdx.y * 128;
  const int col = lane & 15;
  const int rb  = (lane >> 4) << 3;

  v8f acc[8];
#pragma unroll
  for (int gi = 0; gi < 8; ++gi) {
    float bias = bi[g0 + gi * 16 + col] + bh[g0 + gi * 16 + col];
#pragma unroll
    for (int i = 0; i < 8; ++i) acc[gi][i] = bias;
  }

#if HAVE_TDM
  __shared__ __bf16 sA[2][128 * SROW];
  __shared__ __bf16 sB[2][128 * SROW];
  if (wave == 0) {
    tdm_load_2d(sA[0], featsb + (size_t)m0blk * 768, 128, 768);
    tdm_load_2d(sB[0], wih + (size_t)g0 * 768, 128, 768);
  }
  for (int q = 0; q < 24; ++q) {
    const int cur = q & 1;
    if (wave == 0) wait_tensor0();      // staged buffers for step q ready
    __syncthreads();                    // publish; also fences old reads
    if (wave == 0 && q + 1 < 24) {      // kick next K-step into other buffer
      tdm_load_2d(sA[cur ^ 1], featsb + (size_t)m0blk * 768 + (q + 1) * 32, 128, 768);
      tdm_load_2d(sB[cur ^ 1], wih + (size_t)g0 * 768 + (q + 1) * 32, 128, 768);
    }
    v16bf a = frag_ld(&sA[cur][wave * 16 * SROW], SROW);
#pragma unroll
    for (int gi = 0; gi < 8; ++gi) {
      v16bf b = frag_ld(&sB[cur][gi * 16 * SROW], SROW);
      acc[gi] = wmma_bf16(a, b, acc[gi]);
    }
  }
#else
  // fallback: direct global fragment loads, software-pipelined
  v16bf a0 = frag_ld(featsb + (size_t)m0 * 768, 768);
  for (int kk = 0; kk < 768; kk += 32) {
    v16bf a1;
    if (kk + 32 < 768) a1 = frag_ld(featsb + (size_t)m0 * 768 + kk + 32, 768);
#pragma unroll
    for (int gi = 0; gi < 8; ++gi) {
      v16bf b = frag_ld(wih + (size_t)(g0 + gi * 16) * 768 + kk, 768);
      acc[gi] = wmma_bf16(a0, b, acc[gi]);
    }
    a0 = a1;
  }
#endif

  const int batch = m0 >> 9;
  const int lbase = m0 & 511;
#pragma unroll
  for (int gi = 0; gi < 8; ++gi)
#pragma unroll
    for (int i = 0; i < 8; ++i) {
      const int l = lbase + rb + i;
      gout[((size_t)l * 1024 + g0 + gi * 16 + col) * 64 + batch] = (__bf16)acc[gi][i];
    }
}

// ---------------- recurrent LSTM (one WGP-full block per direction) ---------
#define HLD 264
__global__ void __launch_bounds__(1024)
k_lstm(const __bf16* __restrict__ gx_all, const __bf16* __restrict__ whh_f,
       const __bf16* __restrict__ whh_b, __bf16* __restrict__ out) {
  const int dir = blockIdx.x;
  const __bf16* gx  = gx_all + (size_t)dir * 512 * 64 * 1024;
  const __bf16* whh = dir ? whh_b : whh_f;
  const int coloff = dir ? 256 : 0;

  __shared__ __bf16 hls[64 * HLD];

  const int wave = threadIdx.x >> 5;
  const int lane = threadIdx.x & 31;
  const int wq = wave & 15;
  const int wm = wave >> 4;
  const int j0 = wq * 16;
  const int col = lane & 15;
  const int rb  = (lane >> 4) << 3;

  for (int i = threadIdx.x; i < 64 * HLD; i += blockDim.x) hls[i] = (__bf16)0.f;
  float cst[2][8];
#pragma unroll
  for (int mi = 0; mi < 2; ++mi)
#pragma unroll
    for (int i = 0; i < 8; ++i) cst[mi][i] = 0.f;
  __syncthreads();

  for (int t = 0; t < 512; ++t) {
    const int l = dir ? (511 - t) : t;
    const __bf16* gxt = gx + (size_t)l * 64 * 1024;
    if (t + 1 < 512) {                  // warm L2 with next step's 128KB slab
      const int ln = dir ? (510 - t) : (t + 1);
      __builtin_prefetch(gx + (size_t)ln * 64 * 1024 + threadIdx.x * 64, 0, 3);
    }

    // issue gx loads up front; consumed only after the WMMA chain (hidden)
    v8bf gxv[2][4];
#pragma unroll
    for (int mi = 0; mi < 2; ++mi) {
      const int b0 = (wm * 2 + mi) * 16;
#pragma unroll
      for (int gt = 0; gt < 4; ++gt)
        gxv[mi][gt] = *(const v8bf*)(gxt + ((size_t)(gt * 256 + j0 + col) * 64 + b0 + rb));
    }

    v8f acc[2][4];
#pragma unroll
    for (int mi = 0; mi < 2; ++mi)
#pragma unroll
      for (int gt = 0; gt < 4; ++gt)
#pragma unroll
        for (int i = 0; i < 8; ++i) acc[mi][gt][i] = 0.f;

    // K loop over h @ w_hh^T, B-fragments double-buffered
    v16bf bcur[4];
#pragma unroll
    for (int gt = 0; gt < 4; ++gt)
      bcur[gt] = frag_ld(whh + (size_t)(gt * 256 + j0) * 256, 256);
    for (int kk = 0; kk < 256; kk += 32) {
      v16bf bnxt[4];
      if (kk + 32 < 256) {
#pragma unroll
        for (int gt = 0; gt < 4; ++gt)
          bnxt[gt] = frag_ld(whh + (size_t)(gt * 256 + j0) * 256 + kk + 32, 256);
      }
#pragma unroll
      for (int mi = 0; mi < 2; ++mi) {
        v16bf a = frag_ld(hls + (wm * 2 + mi) * 16 * HLD + kk, HLD);
#pragma unroll
        for (int gt = 0; gt < 4; ++gt)
          acc[mi][gt] = wmma_bf16(a, bcur[gt], acc[mi][gt]);
      }
#pragma unroll
      for (int gt = 0; gt < 4; ++gt) bcur[gt] = bnxt[gt];
    }
    __syncthreads();                    // all reads of h done before overwrite

#pragma unroll
    for (int mi = 0; mi < 2; ++mi) {
      const int b0 = (wm * 2 + mi) * 16;
#pragma unroll
      for (int i = 0; i < 8; ++i) {
        const float ig = 1.f / (1.f + __expf(-(acc[mi][0][i] + (float)gxv[mi][0][i])));
        const float fg = 1.f / (1.f + __expf(-(acc[mi][1][i] + (float)gxv[mi][1][i])));
        const float gg = tanhf(acc[mi][2][i] + (float)gxv[mi][2][i]);
        const float og = 1.f / (1.f + __expf(-(acc[mi][3][i] + (float)gxv[mi][3][i])));
        const float cs = fg * cst[mi][i] + ig * gg;
        const float h  = og * tanhf(cs);
        cst[mi][i] = cs;
        hls[(b0 + rb + i) * HLD + j0 + col] = (__bf16)h;
      }
    }
    __syncthreads();                    // h published

    // coalesced writeback of h to lstm_out (lanes -> consecutive j)
    for (int idx = threadIdx.x; idx < 64 * 256; idx += 1024) {
      const int b = idx >> 8, j = idx & 255;
      out[((size_t)b * 512 + l) * 512 + coloff + j] = hls[b * HLD + j];
    }
  }
}

// ---------------- emissions GEMM (K=512, N padded to 16) --------------------
__global__ void __launch_bounds__(256)
k_emis(const __bf16* __restrict__ lstm, const __bf16* __restrict__ fcw,
       const float* __restrict__ fcb, float* __restrict__ emis) {
  const int wave = threadIdx.x >> 5;
  const int lane = threadIdx.x & 31;
  const int r0 = (blockIdx.x * 8 + wave) * 16;
  const int col = lane & 15;
  const int rb  = (lane >> 4) << 3;
  v8f acc;
#pragma unroll
  for (int i = 0; i < 8; ++i) acc[i] = 0.f;
  v16bf a0 = frag_ld(lstm + (size_t)r0 * 512, 512);
  v16bf b0 = frag_ld(fcw, 512);
  for (int kk = 0; kk < 512; kk += 32) {
    v16bf a1, b1;
    if (kk + 32 < 512) {
      a1 = frag_ld(lstm + (size_t)r0 * 512 + kk + 32, 512);
      b1 = frag_ld(fcw + kk + 32, 512);
    }
    acc = wmma_bf16(a0, b0, acc);
    a0 = a1; b0 = b1;
  }
  if (col < 9) {
    const float bias = fcb[col];
#pragma unroll
    for (int i = 0; i < 8; ++i)
      emis[(size_t)(r0 + rb + i) * 9 + col] = acc[i] + bias;
  }
}

// ---------------- CRF forward + Viterbi scores (one wave32 per batch) -------
__global__ void k_crf(const float* __restrict__ emis, const int* __restrict__ mask,
                      const int* __restrict__ labels, const float* __restrict__ start_t,
                      const float* __restrict__ end_t, const float* __restrict__ trans,
                      unsigned char* __restrict__ bp, float* __restrict__ llh,
                      int* __restrict__ lastTag) {
  const int b = blockIdx.x;
  const int j = threadIdx.x;
  const float NEG = -1e30f;
  float trc[9];
#pragma unroll
  for (int i = 0; i < 9; ++i) trc[i] = (j < 9) ? trans[i * 9 + j] : 0.f;

  const float* em = emis + (size_t)b * 512 * 9;
  float score = (j < 9) ? (start_t[j] + em[j]) : NEG;
  float vsc = score;

  int prev = labels[b * 512]; if (prev == -100) prev = 0;
  float num = start_t[prev] + em[prev];
  int msum = mask[b * 512] ? 1 : 0;

  for (int t = 1; t < 512; ++t) {
    const int mt = mask[b * 512 + t];
    const float* emt = em + t * 9;
    const float e = (j < 9) ? emt[j] : NEG;
    float mx = NEG;
#pragma unroll
    for (int i = 0; i < 9; ++i) mx = fmaxf(mx, __shfl(score, i, 32) + trc[i]);
    float s = 0.f;
#pragma unroll
    for (int i = 0; i < 9; ++i) s += __expf(__shfl(score, i, 32) + trc[i] - mx);
    const float nsc = mx + __logf(s) + e;
    float best = NEG; int bi = 0;
#pragma unroll
    for (int i = 0; i < 9; ++i) {
      float v = __shfl(vsc, i, 32) + trc[i];
      if (v > best) { best = v; bi = i; }
    }
    const float nvs = best + e;
    if (j < 9) bp[((size_t)b * 512 + t) * 9 + j] = (unsigned char)bi;
    if (mt) { score = nsc; vsc = nvs; }
    if (j == 0) {
      int tg = labels[b * 512 + t]; if (tg == -100) tg = 0;
      if (mt) num += trans[prev * 9 + tg] + emt[tg];
      prev = tg;
      msum += mt ? 1 : 0;
    }
  }
  float mx2 = NEG;
#pragma unroll
  for (int i = 0; i < 9; ++i) mx2 = fmaxf(mx2, __shfl(score, i, 32) + end_t[i]);
  float s2 = 0.f;
#pragma unroll
  for (int i = 0; i < 9; ++i) s2 += __expf(__shfl(score, i, 32) + end_t[i] - mx2);
  const float denom = mx2 + __logf(s2);
  float bestv = NEG; int bti = 0;
#pragma unroll
  for (int i = 0; i < 9; ++i) {
    float v = __shfl(vsc, i, 32) + end_t[i];
    if (v > bestv) { bestv = v; bti = i; }
  }
  if (j == 0) {
    int li = msum - 1; if (li < 0) li = 0;
    int lt = labels[b * 512 + li]; if (lt == -100) lt = 0;
    num += end_t[lt];
    llh[b] = num - denom;
    lastTag[b] = bti;
  }
}

// ---------------- Viterbi backtrace + deterministic loss --------------------
__global__ void k_viterbi_bt(const unsigned char* __restrict__ bp, const int* __restrict__ mask,
                             const int* __restrict__ lastTag, const float* __restrict__ llh,
                             float* __restrict__ loss_out, int* __restrict__ preds) {
  const int b = threadIdx.x;
  if (b < 64) {
    int tag = lastTag[b];
    preds[b * 512 + 511] = mask[b * 512 + 511] ? tag : -100;
    for (int t = 511; t >= 1; --t) {
      const int pv = bp[((size_t)b * 512 + t) * 9 + tag];
      if (mask[b * 512 + t]) tag = pv;
      preds[b * 512 + t - 1] = mask[b * 512 + t - 1] ? tag : -100;
    }
  }
  if (threadIdx.x == 0) {
    float s = 0.f;
    for (int i = 0; i < 64; ++i) s += llh[i];
    loss_out[0] = -s / 64.f;
  }
}

// ---------------- host-side orchestration ----------------------------------
extern "C" void kernel_launch(void* const* d_in, const int* in_sizes, int n_in,
                              void* d_out, int out_size, void* d_ws, size_t ws_size,
                              hipStream_t stream) {
  (void)in_sizes; (void)n_in; (void)out_size; (void)ws_size;
  const float* feats   = (const float*)d_in[0];
  const int*   mask    = (const int*)d_in[1];
  const int*   labels  = (const int*)d_in[2];
  const float* w_ih_f  = (const float*)d_in[3];
  const float* w_hh_f  = (const float*)d_in[4];
  const float* b_ih_f  = (const float*)d_in[5];
  const float* b_hh_f  = (const float*)d_in[6];
  const float* w_ih_b  = (const float*)d_in[7];
  const float* w_hh_b  = (const float*)d_in[8];
  const float* b_ih_b  = (const float*)d_in[9];
  const float* b_hh_b  = (const float*)d_in[10];
  const float* fc_w    = (const float*)d_in[11];
  const float* fc_b    = (const float*)d_in[12];
  const float* start_t = (const float*)d_in[13];
  const float* end_t   = (const float*)d_in[14];
  const float* trans   = (const float*)d_in[15];

  char* ws = (char*)d_ws;
  size_t off = 0;
  auto alloc = [&](size_t bytes) -> void* {
    void* p = ws + off;
    off = (off + bytes + 255) & ~(size_t)255;
    return p;
  };
  __bf16* featsb = (__bf16*)alloc((size_t)64 * 512 * 768 * 2);
  __bf16* wihf   = (__bf16*)alloc((size_t)1024 * 768 * 2);
  __bf16* wihb   = (__bf16*)alloc((size_t)1024 * 768 * 2);
  __bf16* whhf   = (__bf16*)alloc((size_t)1024 * 256 * 2);
  __bf16* whhb   = (__bf16*)alloc((size_t)1024 * 256 * 2);
  __bf16* gx     = (__bf16*)alloc((size_t)2 * 512 * 64 * 1024 * 2);
  __bf16* lstmo  = (__bf16*)alloc((size_t)64 * 512 * 512 * 2);
  __bf16* fcwb   = (__bf16*)alloc((size_t)16 * 512 * 2);
  float*  emis   = (float*)alloc((size_t)64 * 512 * 9 * 4);
  unsigned char* bp = (unsigned char*)alloc((size_t)64 * 512 * 9);
  float*  llh    = (float*)alloc(64 * 4);
  int*    lastT  = (int*)alloc(64 * 4);

  k_cvt_bf16<<<2048, 256, 0, stream>>>(feats, featsb, 64 * 512 * 768);
  k_cvt_bf16<<<768, 256, 0, stream>>>(w_ih_f, wihf, 1024 * 768);
  k_cvt_bf16<<<768, 256, 0, stream>>>(w_ih_b, wihb, 1024 * 768);
  k_cvt_bf16<<<256, 256, 0, stream>>>(w_hh_f, whhf, 1024 * 256);
  k_cvt_bf16<<<256, 256, 0, stream>>>(w_hh_b, whhb, 1024 * 256);
  k_pack_fcw<<<32, 256, 0, stream>>>(fc_w, fcwb);

  dim3 g1(256, 8, 2);
  k_gemm_gx<<<g1, 256, 0, stream>>>(featsb, wihf, wihb, b_ih_f, b_hh_f,
                                    b_ih_b, b_hh_b, gx);
  k_lstm<<<2, 1024, 0, stream>>>(gx, whhf, whhb, lstmo);
  k_emis<<<256, 256, 0, stream>>>(lstmo, fcwb, fc_b, emis);
  k_crf<<<64, 32, 0, stream>>>(emis, mask, labels, start_t, end_t, trans,
                               bp, llh, lastT);
  k_viterbi_bt<<<1, 64, 0, stream>>>(bp, mask, lastT, llh,
                                     (float*)d_out, ((int*)d_out) + 1);
}
